// _HeteroGATHead_71597104824560
// MI455X (gfx1250) — matrix operand order repaired and checked by reading, not compiled, not run on hardware
//
#include <hip/hip_runtime.h>
#include <hip/hip_bf16.h>

#define NG   4096
#define NC_  4096
#define FIN  512
#define DD   256
#define ALPHA 0.2f

typedef __attribute__((ext_vector_type(16))) __bf16 v16bf;
typedef __attribute__((ext_vector_type(8)))  float  v8f;

union ABFrag { v16bf v; uint4 q[2]; };

__device__ __forceinline__ unsigned short f2bf(float f) {
  unsigned int u = __float_as_uint(f);
  u += 0x7FFFu + ((u >> 16) & 1u);           // round-to-nearest-even
  return (unsigned short)(u >> 16);
}
__device__ __forceinline__ float leakyf(float x) { return x >= 0.f ? x : ALPHA * x; }
__device__ __forceinline__ v8f zero8() { v8f z = {0.f,0.f,0.f,0.f,0.f,0.f,0.f,0.f}; return z; }

// ---------------- K0: fp32 -> bf16 conversion ----------------
__global__ void k_convert_bf16(const float* __restrict__ src,
                               unsigned short* __restrict__ dst, int n) {
  int i = blockIdx.x * blockDim.x + threadIdx.x;
  if (i < n) dst[i] = f2bf(src[i]);
}

// ---------------- K1: H = X[N,FIN] @ W[D,FIN]^T  (bf16 WMMA, fp32 accum) ---
// Writes Hf32 [N,DD] and HT bf16 [DD,N] (transposed, for attention B-frags).
__global__ __launch_bounds__(128) void k_h_gemm(
    const unsigned short* __restrict__ Xbf, const unsigned short* __restrict__ Wbf,
    float* __restrict__ Hf, unsigned short* __restrict__ HTbf, int Nrows)
{
  const int tid  = threadIdx.x;
  const int wave = tid >> 5;
  const int lane = tid & 31;
  const int m0   = blockIdx.x * 16;
  const int cb0  = wave * 64;

  v8f acc[4];
  #pragma unroll
  for (int i = 0; i < 4; ++i) acc[i] = zero8();

  // A-frag (16x32 bf16): lanes 0-15 row=lane, K{kb..kb+15};
  // lanes 16-31 row=lane-16, K{kb+16..kb+31} (contiguous, matches B)
  const int rA   = lane & 15;
  const int kofs = (lane >> 4) * 16;

  for (int kb = 0; kb < FIN; kb += 32) {
    ABFrag a;
    const unsigned short* ap = Xbf + (size_t)(m0 + rA) * FIN + kb + kofs;
    a.q[0] = *(const uint4*)(ap);
    a.q[1] = *(const uint4*)(ap + 8);
    #pragma unroll
    for (int ct = 0; ct < 4; ++ct) {
      // B-frag (32x16): lane col n = lane&15, K = kb + (lane>>4)*16 .. +15 contiguous
      int n = cb0 + ct * 16 + (lane & 15);
      const unsigned short* bp = Wbf + (size_t)n * FIN + kb + (lane >> 4) * 16;
      ABFrag b;
      b.q[0] = *(const uint4*)(bp);
      b.q[1] = *(const uint4*)(bp + 8);
      acc[ct] = __builtin_amdgcn_wmma_f32_16x16x32_bf16(
          false, a.v, false, b.v, (short)0, acc[ct], false, false);
    }
  }
  // C/D layout: VGPR v, lanes 0-15 -> M=v, N=lane; lanes 16-31 -> M=v+8, N=lane-16
  #pragma unroll
  for (int ct = 0; ct < 4; ++ct) {
    int n  = cb0 + ct * 16 + (lane & 15);
    int mo = (lane >> 4) ? 8 : 0;
    #pragma unroll
    for (int v = 0; v < 8; ++v) {
      int m = m0 + v + mo;
      float val = acc[ct][v];
      Hf[(size_t)m * DD + n]          = val;
      HTbf[(size_t)n * Nrows + m]     = f2bf(val);
    }
  }
}

// ---------------- K2: per-row dot products + sigmoid gate -----------------
__global__ __launch_bounds__(256) void k_svec(
    const float* __restrict__ H, int N,
    const float* __restrict__ aA, const float* __restrict__ aB,
    const float* __restrict__ aC, const float* __restrict__ aD,
    const float* __restrict__ gw, const float* __restrict__ gb,
    float* __restrict__ sA, float* __restrict__ sB,
    float* __restrict__ sC, float* __restrict__ sD,
    float* __restrict__ gamma)
{
  int wave = threadIdx.x >> 5, lane = threadIdx.x & 31;
  int row = blockIdx.x * 8 + wave;
  if (row >= N) return;
  const float* h = H + (size_t)row * DD;
  float dA = 0.f, dB = 0.f, dC = 0.f, dDv = 0.f, dG = 0.f;
  for (int t = lane; t < DD; t += 32) {
    float hv = h[t];
    dA += hv * aA[t]; dB += hv * aB[t]; dC += hv * aC[t];
    dDv += hv * aD[t]; dG += hv * gw[t];
  }
  #pragma unroll
  for (int m = 16; m >= 1; m >>= 1) {
    dA  += __shfl_xor(dA, m, 32);  dB += __shfl_xor(dB, m, 32);
    dC  += __shfl_xor(dC, m, 32);  dDv += __shfl_xor(dDv, m, 32);
    dG  += __shfl_xor(dG, m, 32);
  }
  if (lane == 0) {
    sA[row] = dA; sB[row] = dB; sC[row] = dC; sD[row] = dDv;
    gamma[row] = 1.f / (1.f + __expf(-(dG + gb[0])));
  }
}

// ---------------- K2b: global max of each s2 vector -----------------------
__global__ __launch_bounds__(256) void k_maxred(
    const float* p0, const float* p1, const float* p2, const float* p3,
    int n, float* __restrict__ m2)
{
  const float* p = (blockIdx.x == 0) ? p0 : (blockIdx.x == 1) ? p1
                 : (blockIdx.x == 2) ? p2 : p3;
  float mv = -3.4e38f;
  for (int i = threadIdx.x; i < n; i += 256) mv = fmaxf(mv, p[i]);
  __shared__ float red[256];
  red[threadIdx.x] = mv;
  __syncthreads();
  for (int s = 128; s > 0; s >>= 1) {
    if (threadIdx.x < s) red[threadIdx.x] = fmaxf(red[threadIdx.x], red[threadIdx.x + s]);
    __syncthreads();
  }
  if (threadIdx.x == 0) m2[blockIdx.x] = red[0];
}

// ---------------- K3: fused masked-softmax attention @ Hdst ---------------
// out[16 src rows, DD] = softmax_j(adj ? leaky(s1_i + s2_j) : -inf) @ Hdst
// Single pass: p = exp(e - leaky(s1_i + max_j s2_j)); normalize by row sum.
__global__ __launch_bounds__(128) void k_attn(
    const float* __restrict__ adj, long rstride, long cstride,
    const float* __restrict__ s1, const float* __restrict__ s2,
    const float* __restrict__ m2, int m2idx,
    const unsigned short* __restrict__ HT, int Ndst,
    float* __restrict__ outBuf)
{
  const int tid = threadIdx.x, wave = tid >> 5, lane = tid & 31;
  const int src0 = blockIdx.x * 16;

  __shared__ unsigned short pTile[16][32];   // bf16 P tile (A matrix)
  __shared__ float s1L[16], mL[16], rsumL[16];

  if (tid < 16) {
    float s = s1[src0 + tid];
    s1L[tid] = s;
    mL[tid]  = leakyf(s + m2[m2idx]);
  }
  __syncthreads();

  const int   r   = tid >> 3;          // src row handled by this lane (0..15)
  const int   c0  = (tid & 7) * 4;     // 4 dst cols per lane
  const float s1r = s1L[r];
  const float mr  = mL[r];
  float psum = 0.f;

  v8f acc[4];
  #pragma unroll
  for (int i = 0; i < 4; ++i) acc[i] = zero8();

  const int rA = lane & 15, kofs = (lane >> 4) * 16;

  for (int jb = 0; jb < Ndst; jb += 32) {
    // ---- build P tile (all 128 threads) ----
    const float* arow = adj + (size_t)(src0 + r) * rstride + (size_t)(jb + c0) * cstride;
    float av[4];
    if (cstride == 1) {
      float4 q = *(const float4*)arow;
      av[0] = q.x; av[1] = q.y; av[2] = q.z; av[3] = q.w;
    } else {
      #pragma unroll
      for (int t = 0; t < 4; ++t) av[t] = arow[(size_t)t * cstride];
    }
    if (jb + 32 < Ndst)  // stream next adjacency chunk through the cache early
      __builtin_prefetch(adj + (size_t)(src0 + r) * rstride + (size_t)(jb + 32 + c0) * cstride, 0, 0);

    #pragma unroll
    for (int t = 0; t < 4; ++t) {
      float e = leakyf(s1r + s2[jb + c0 + t]);
      float p = (av[t] > 0.f) ? exp2f((e - mr) * 1.44269504f) : 0.f;
      psum += p;
      pTile[r][c0 + t] = f2bf(p);
    }
    __syncthreads();

    // ---- WMMA: each wave owns 64 output columns ----
    ABFrag a;
    a.q[0] = *(const uint4*)&pTile[rA][kofs];
    a.q[1] = *(const uint4*)&pTile[rA][kofs + 8];
    long koff = (long)jb + (lane >> 4) * 16;
    #pragma unroll
    for (int ct = 0; ct < 4; ++ct) {
      int n = wave * 64 + ct * 16 + (lane & 15);
      const unsigned short* bp = HT + (size_t)n * Ndst + koff;
      ABFrag b;
      b.q[0] = *(const uint4*)(bp);
      b.q[1] = *(const uint4*)(bp + 8);
      acc[ct] = __builtin_amdgcn_wmma_f32_16x16x32_bf16(
          false, a.v, false, b.v, (short)0, acc[ct], false, false);
    }
    __syncthreads();
  }

  // ---- row-sum reduce: 8 consecutive lanes per row (within one wave) ----
  psum += __shfl_xor(psum, 1, 32);
  psum += __shfl_xor(psum, 2, 32);
  psum += __shfl_xor(psum, 4, 32);
  if ((tid & 7) == 0) rsumL[r] = psum;
  __syncthreads();

  #pragma unroll
  for (int ct = 0; ct < 4; ++ct) {
    int n  = wave * 64 + ct * 16 + (lane & 15);
    int mo = (lane >> 4) ? 8 : 0;
    #pragma unroll
    for (int v = 0; v < 8; ++v) {
      int m = v + mo;
      float rs = rsumL[m];
      rs = (rs > 0.f) ? rs : 1.f;
      outBuf[(size_t)(src0 + m) * DD + n] = acc[ct][v] / rs;
    }
  }
}

// ---------------- K4: gated combine + leaky, write d_out ------------------
__global__ void k_combine(
    const float* __restrict__ Hg, const float* __restrict__ Hc,
    const float* __restrict__ bgg, const float* __restrict__ bgc,
    const float* __restrict__ bcc, const float* __restrict__ bcg,
    const float* __restrict__ gg, const float* __restrict__ gc,
    float* __restrict__ out)
{
  size_t i = (size_t)blockIdx.x * blockDim.x + threadIdx.x;
  size_t half = (size_t)NG * DD;
  if (i < half) {
    size_t row = i / DD;
    out[i] = leakyf(Hg[i] + bgg[i] + gg[row] * bgc[i]);
  } else if (i < 2 * half) {
    size_t j = i - half, row = j / DD;
    out[i] = leakyf(Hc[j] + bcc[j] + gc[row] * bcg[j]);
  }
}

// --------------------------------------------------------------------------
extern "C" void kernel_launch(void* const* d_in, const int* in_sizes, int n_in,
                              void* d_out, int out_size, void* d_ws, size_t ws_size,
                              hipStream_t stream)
{
  (void)in_sizes; (void)n_in; (void)out_size; (void)ws_size;
  const float* gene_x   = (const float*)d_in[0];
  const float* cell_x   = (const float*)d_in[1];
  const float* gene_adj = (const float*)d_in[2];
  const float* cell_adj = (const float*)d_in[3];
  const float* gc_adj   = (const float*)d_in[4];
  const float* Wg       = (const float*)d_in[5];
  const float* Wc       = (const float*)d_in[6];
  const float* a_gg     = (const float*)d_in[7];
  const float* a_gc     = (const float*)d_in[8];
  const float* a_cc     = (const float*)d_in[9];
  const float* a_cg     = (const float*)d_in[10];
  const float* gate_g_w = (const float*)d_in[11];
  const float* gate_g_b = (const float*)d_in[12];
  const float* gate_c_w = (const float*)d_in[13];
  const float* gate_c_b = (const float*)d_in[14];
  float* out = (float*)d_out;

  char* ws = (char*)d_ws;
  size_t off = 0;
  auto alloc = [&](size_t bytes) -> char* {
    char* p = ws + off;
    off += (bytes + 255) & ~(size_t)255;
    return p;
  };
  unsigned short* Xg_bf = (unsigned short*)alloc((size_t)NG * FIN * 2);
  unsigned short* Xc_bf = (unsigned short*)alloc((size_t)NC_ * FIN * 2);
  unsigned short* Wg_bf = (unsigned short*)alloc((size_t)DD * FIN * 2);
  unsigned short* Wc_bf = (unsigned short*)alloc((size_t)DD * FIN * 2);
  float*          Hg_f  = (float*)alloc((size_t)NG * DD * 4);
  float*          Hc_f  = (float*)alloc((size_t)NC_ * DD * 4);
  unsigned short* HgT   = (unsigned short*)alloc((size_t)DD * NG * 2);
  unsigned short* HcT   = (unsigned short*)alloc((size_t)DD * NC_ * 2);
  float* s_gg1 = (float*)alloc(NG * 4);
  float* s_gg2 = (float*)alloc(NG * 4);
  float* s_gc1 = (float*)alloc(NG * 4);
  float* s_gc2 = (float*)alloc(NC_ * 4);
  float* s_cc1 = (float*)alloc(NC_ * 4);
  float* s_cc2 = (float*)alloc(NC_ * 4);
  float* s_cg1 = (float*)alloc(NC_ * 4);
  float* s_cg2 = (float*)alloc(NG * 4);
  float* gam_g = (float*)alloc(NG * 4);
  float* gam_c = (float*)alloc(NC_ * 4);
  float* m2    = (float*)alloc(4 * 4);
  float* bGG   = (float*)alloc((size_t)NG * DD * 4);
  float* bGC   = (float*)alloc((size_t)NG * DD * 4);
  float* bCC   = (float*)alloc((size_t)NC_ * DD * 4);
  float* bCG   = (float*)alloc((size_t)NC_ * DD * 4);

  // K0: conversions
  k_convert_bf16<<<(NG * FIN + 255) / 256, 256, 0, stream>>>(gene_x, Xg_bf, NG * FIN);
  k_convert_bf16<<<(NC_ * FIN + 255) / 256, 256, 0, stream>>>(cell_x, Xc_bf, NC_ * FIN);
  k_convert_bf16<<<(DD * FIN + 255) / 256, 256, 0, stream>>>(Wg, Wg_bf, DD * FIN);
  k_convert_bf16<<<(DD * FIN + 255) / 256, 256, 0, stream>>>(Wc, Wc_bf, DD * FIN);

  // K1: feature GEMMs
  k_h_gemm<<<NG / 16, 128, 0, stream>>>(Xg_bf, Wg_bf, Hg_f, HgT, NG);
  k_h_gemm<<<NC_ / 16, 128, 0, stream>>>(Xc_bf, Wc_bf, Hc_f, HcT, NC_);

  // K2: attention scalars + gates
  k_svec<<<NG / 8, 256, 0, stream>>>(Hg_f, NG, a_gg, a_gg + DD, a_gc, a_cg + DD,
                                     gate_g_w, gate_g_b,
                                     s_gg1, s_gg2, s_gc1, s_cg2, gam_g);
  k_svec<<<NC_ / 8, 256, 0, stream>>>(Hc_f, NC_, a_cc, a_cc + DD, a_gc + DD, a_cg,
                                      gate_c_w, gate_c_b,
                                      s_cc1, s_cc2, s_gc2, s_cg1, gam_c);

  // K2b: global maxes of s2 vectors (softmax stabilizer bound)
  k_maxred<<<4, 256, 0, stream>>>(s_gg2, s_gc2, s_cc2, s_cg2, NG, m2);

  // K3: four fused attention blocks
  k_attn<<<NG / 16, 128, 0, stream>>>(gene_adj, (long)NG, 1L,
                                      s_gg1, s_gg2, m2, 0, HgT, NG, bGG);
  k_attn<<<NG / 16, 128, 0, stream>>>(gc_adj, (long)NC_, 1L,
                                      s_gc1, s_gc2, m2, 1, HcT, NC_, bGC);
  k_attn<<<NC_ / 16, 128, 0, stream>>>(cell_adj, (long)NC_, 1L,
                                       s_cc1, s_cc2, m2, 2, HcT, NC_, bCC);
  k_attn<<<NC_ / 16, 128, 0, stream>>>(gc_adj, 1L, (long)NC_,   // transposed view
                                       s_cg1, s_cg2, m2, 3, HgT, NG, bCG);

  // K4: combine
  size_t total = 2 * (size_t)NG * DD;
  k_combine<<<(unsigned)((total + 255) / 256), 256, 0, stream>>>(
      Hg_f, Hc_f, bGG, bGC, bCC, bCG, gam_g, gam_c, out);
}
